// TransformerBlock_3143916061089
// MI455X (gfx1250) — matrix operand order repaired and checked by reading, not compiled, not run on hardware
//
#include <hip/hip_runtime.h>
#include <math.h>

// ---------------------------------------------------------------------------
// Transformer block for MI455X (gfx1250, wave32, WMMA + TDM + async-LDS).
// ---------------------------------------------------------------------------

typedef __attribute__((ext_vector_type(8)))  float  v8f;
typedef __attribute__((ext_vector_type(8)))  __bf16 bf16x8;
typedef __attribute__((ext_vector_type(16))) __bf16 bf16x16;
typedef __attribute__((ext_vector_type(4)))  unsigned int u32x4;
typedef __attribute__((ext_vector_type(4)))  int i32x4;
typedef __attribute__((ext_vector_type(8)))  int i32x8;

static __device__ __forceinline__ v8f vzero8() {
  v8f v = {0.f,0.f,0.f,0.f,0.f,0.f,0.f,0.f};
  return v;
}

// ---- CDNA5 async global->LDS (ASYNCcnt) -----------------------------------
// Inline asm: VGLOBAL async form "global_load_async_to_lds_b128 vdst, vaddr, off"
// (VDST = LDS byte address VGPR, VADDR = 64-bit global address pair).
static __device__ __forceinline__ void async_copy_b128(void* lds, const void* g) {
  unsigned l = (unsigned)(uintptr_t)lds;  // low 32 bits of LDS-aperture addr = LDS offset
  unsigned long long ga = (unsigned long long)(uintptr_t)g;
  asm volatile("global_load_async_to_lds_b128 %0, %1, off"
               :: "v"(l), "v"(ga) : "memory");
}

static __device__ __forceinline__ void wait_asynccnt0() {
#if __has_builtin(__builtin_amdgcn_s_wait_asynccnt)
  __builtin_amdgcn_s_wait_asynccnt(0);
#else
  asm volatile("s_wait_asynccnt 0x0" ::: "memory");
#endif
}

static __device__ __forceinline__ void wait_tensorcnt0() {
  __builtin_amdgcn_s_wait_tensorcnt(0);
}

// ---- CDNA5 Tensor Data Mover: 2-D tile of 2-byte elements -> LDS ----------
// Descriptor packing per CDNA5 ISA 8.3/8.4 (group0 128b, group1 256b).
static __device__ __forceinline__ void tdm_load_2d_bf16(
    void* lds, const void* g, unsigned tile_w, unsigned tile_h,
    unsigned row_stride_elems) {
  unsigned long long ga = (unsigned long long)(uintptr_t)g;
  u32x4 g0;
  g0[0] = 1u;                                   // count=1 (valid user D#)
  g0[1] = (unsigned)(uintptr_t)lds;             // lds_addr (bytes)
  g0[2] = (unsigned)(ga & 0xffffffffu);         // global_addr[31:0]
  g0[3] = (unsigned)((ga >> 32) & 0x1ffffffu)   // global_addr[56:32]
          | (2u << 30);                         // type=2 ("image")
  i32x8 g1;
  g1[0] = (int)(1u << 16);                      // data_size=1 -> 2 bytes
  g1[1] = (int)(tile_w << 16);                  // tensor_dim0[15:0]
  g1[2] = (int)((tile_w >> 16) | (tile_h << 16));   // dim0 hi | dim1 lo
  g1[3] = (int)((tile_h >> 16) | (tile_w << 16));   // dim1 hi | tile_dim0
  g1[4] = (int)(tile_h & 0xffffu);              // tile_dim1 (tile_dim2=0)
  g1[5] = (int)row_stride_elems;                // tensor_dim0_stride[31:0]
  g1[6] = 0;                                    // stride hi | dim1_stride lo
  g1[7] = 0;
  i32x4 z4 = {0, 0, 0, 0};
#if defined(__clang_major__) && (__clang_major__ >= 23)
  i32x8 z8 = {0, 0, 0, 0, 0, 0, 0, 0};
  __builtin_amdgcn_tensor_load_to_lds(g0, g1, z4, z4, z8, 0);
#else
  __builtin_amdgcn_tensor_load_to_lds(g0, g1, z4, z4, 0);
#endif
}

// ---- WMMA fragment loaders (ISA 7.12.2 layouts) ---------------------------
static __device__ __forceinline__ bf16x16 frag_a(const __bf16* base, int stride, int lane) {
  const int m  = lane & 15;
  const int kh = lane >> 4;
  bf16x8 lo = *(const bf16x8*)(base + m * stride + 8 * kh);
  bf16x8 hi = *(const bf16x8*)(base + m * stride + 16 + 8 * kh);
  return __builtin_shufflevector(lo, hi, 0,1,2,3,4,5,6,7,8,9,10,11,12,13,14,15);
}

static __device__ __forceinline__ bf16x16 frag_bt(const __bf16* base, int stride, int lane) {
  const int n  = lane & 15;
  const int kh = lane >> 4;
  bf16x8 lo = *(const bf16x8*)(base + n * stride + 8 * kh);
  bf16x8 hi = *(const bf16x8*)(base + n * stride + 16 + 8 * kh);
  return __builtin_shufflevector(lo, hi, 0,1,2,3,4,5,6,7,8,9,10,11,12,13,14,15);
}

static __device__ __forceinline__ v8f wmma_bf16(bf16x16 a, bf16x16 b, v8f c) {
  return __builtin_amdgcn_wmma_f32_16x16x32_bf16(false, a, false, b, (short)0, c, false, false);
}

// ---------------------------------------------------------------------------
// Weight convert + transpose: Wt[n*K + k] = (bf16) W[k*N + n]
// ---------------------------------------------------------------------------
__global__ __launch_bounds__(256) void transpose_cvt_kernel(
    const float* __restrict__ W, __bf16* __restrict__ Wt, int K, int N) {
  size_t idx = (size_t)blockIdx.x * 256 + threadIdx.x;
  if (idx < (size_t)K * N) {
    int k = (int)(idx / N);
    int n = (int)(idx % N);
    Wt[(size_t)n * K + k] = (__bf16)W[idx];
  }
}

// ---------------------------------------------------------------------------
// LayerNorm (C=1024, eps=1e-3): one block per row, f32 in -> bf16 out
// ---------------------------------------------------------------------------
__global__ __launch_bounds__(256) void layernorm_bf16_kernel(
    const float* __restrict__ x, const float* __restrict__ g,
    const float* __restrict__ beta, __bf16* __restrict__ out) {
  const int C = 1024;
  const int row  = blockIdx.x;
  const int tid  = threadIdx.x;
  const int lane = tid & 31;
  const int wid  = tid >> 5;

  const float4 v = ((const float4*)(x + (size_t)row * C))[tid];
  float s = v.x + v.y + v.z + v.w;
#pragma unroll
  for (int off = 16; off > 0; off >>= 1) s += __shfl_xor(s, off);

  __shared__ float red[8];
  __shared__ float statMu;
  __shared__ float statRs;
  if (lane == 0) red[wid] = s;
  __syncthreads();
  if (tid == 0) {
    float t = 0.f;
#pragma unroll
    for (int i = 0; i < 8; ++i) t += red[i];
    statMu = t * (1.0f / C);
  }
  __syncthreads();
  const float mu = statMu;
  const float dx = v.x - mu, dy = v.y - mu, dz = v.z - mu, dw = v.w - mu;
  float s2 = dx * dx + dy * dy + dz * dz + dw * dw;
#pragma unroll
  for (int off = 16; off > 0; off >>= 1) s2 += __shfl_xor(s2, off);
  __syncthreads();
  if (lane == 0) red[wid] = s2;
  __syncthreads();
  if (tid == 0) {
    float t = 0.f;
#pragma unroll
    for (int i = 0; i < 8; ++i) t += red[i];
    statRs = rsqrtf(t * (1.0f / C) + 1e-3f);
  }
  __syncthreads();
  const float rs = statRs;
  const int base = tid * 4;
  __bf16* o = out + (size_t)row * C + base;
  o[0] = (__bf16)(dx * rs * g[base + 0] + beta[base + 0]);
  o[1] = (__bf16)(dy * rs * g[base + 1] + beta[base + 1]);
  o[2] = (__bf16)(dz * rs * g[base + 2] + beta[base + 2]);
  o[3] = (__bf16)(dw * rs * g[base + 3] + beta[base + 3]);
}

// ---------------------------------------------------------------------------
// Tiled WMMA GEMM with double-buffered TDM staging.
// C[M,N] = A[M,K] x B[K,N] (B given transposed: Bt[N,K]).
// Block = 256 thr (8 waves), tile 128x128, BK=32; wave tile 32x64 (2x4 WMMA).
// Wave 0 drives the Tensor Data Mover: while all waves compute on buffer b,
// TDM streams the next K-tile into buffer b^1 (overlap, TENSORcnt-synced).
// MODE 0: bf16(acc+bias)  MODE 1: bf16(relu(acc+bias))
// MODE 2: f32 acc+bias+resid  MODE 3: QKV scatter (Q*0.125, V transposed)
// ---------------------------------------------------------------------------
template <int MODE>
__global__ __launch_bounds__(256) void gemm_wmma_kernel(
    const __bf16* __restrict__ A, const __bf16* __restrict__ Bt,
    int M, int N, int K,
    const float* __restrict__ bias0, const float* __restrict__ bias1,
    const float* __restrict__ bias2, const float* __restrict__ resid,
    float* __restrict__ outF, __bf16* __restrict__ outB0,
    __bf16* __restrict__ outB1, __bf16* __restrict__ outB2, int T) {
  __shared__ __bf16 As[2][128 * 32];
  __shared__ __bf16 Bs[2][128 * 32];

  const int tid  = threadIdx.x;
  const int lane = tid & 31;
  const int wid  = tid >> 5;
  const int wm   = wid & 3;
  const int wn   = wid >> 2;
  const int m0   = blockIdx.y * 128;
  const int n0   = blockIdx.x * 128;

  v8f acc[2][4];
#pragma unroll
  for (int i = 0; i < 2; ++i)
#pragma unroll
    for (int j = 0; j < 4; ++j) acc[i][j] = vzero8();

  const int nIter = K / 32;
  if (wid == 0) {  // prologue DMA: tile 0 into buffer 0
    tdm_load_2d_bf16(&As[0][0], A  + (size_t)m0 * K, 32, 128, (unsigned)K);
    tdm_load_2d_bf16(&Bs[0][0], Bt + (size_t)n0 * K, 32, 128, (unsigned)K);
    wait_tensorcnt0();
  }
  __syncthreads();

  for (int it = 0; it < nIter; ++it) {
    const int buf = it & 1;
    if (wid == 0 && it + 1 < nIter) {  // prefetch next tile via TDM
      const int k0n = (it + 1) * 32;
      tdm_load_2d_bf16(&As[buf ^ 1][0], A  + (size_t)m0 * K + k0n, 32, 128, (unsigned)K);
      tdm_load_2d_bf16(&Bs[buf ^ 1][0], Bt + (size_t)n0 * K + k0n, 32, 128, (unsigned)K);
    }

    bf16x16 af[2], bfr[4];
#pragma unroll
    for (int i = 0; i < 2; ++i) af[i]  = frag_a (&As[buf][(wm * 32 + i * 16) * 32], 32, lane);
#pragma unroll
    for (int j = 0; j < 4; ++j) bfr[j] = frag_bt(&Bs[buf][(wn * 64 + j * 16) * 32], 32, lane);
#pragma unroll
    for (int i = 0; i < 2; ++i)
#pragma unroll
      for (int j = 0; j < 4; ++j) acc[i][j] = wmma_bf16(af[i], bfr[j], acc[i][j]);

    if (wid == 0) wait_tensorcnt0();
    __syncthreads();
  }

  const int nl = lane & 15;
  const int kh = lane >> 4;
#pragma unroll
  for (int j = 0; j < 4; ++j) {
    const int n = n0 + wn * 64 + j * 16 + nl;
#pragma unroll
    for (int i = 0; i < 2; ++i) {
#pragma unroll
      for (int r = 0; r < 8; ++r) {
        const int m = m0 + wm * 32 + i * 16 + r + 8 * kh;
        const float v = acc[i][j][r];
        if constexpr (MODE == 0) {
          outB0[(size_t)m * N + n] = (__bf16)(v + bias0[n]);
        } else if constexpr (MODE == 1) {
          float t = v + bias0[n];
          outB0[(size_t)m * N + n] = (__bf16)(t > 0.f ? t : 0.f);
        } else if constexpr (MODE == 2) {
          outF[(size_t)m * N + n] = v + bias0[n] + resid[(size_t)m * N + n];
        } else {  // QKV scatter
          const int mat = n >> 10;
          const int jj  = n & 1023;
          const int h   = jj >> 6;
          const int d   = jj & 63;
          const int b   = m >> 11;
          const int t   = m & 2047;
          if (mat == 0) {
            outB0[(((size_t)(b * 16 + h)) * T + t) * 64 + d] =
                (__bf16)((v + bias0[jj]) * 0.125f);
          } else if (mat == 1) {
            outB1[(((size_t)(b * 16 + h)) * T + t) * 64 + d] = (__bf16)(v + bias1[jj]);
          } else {
            outB2[(((size_t)(b * 16 + h)) * 64 + d) * T + t] = (__bf16)(v + bias2[jj]);
          }
        }
      }
    }
  }
}

// ---------------------------------------------------------------------------
// Flash-style causal attention. Grid: (T/128, B*H). Block = 256 (8 waves).
// K / V^T tiles staged with async global->LDS (ASYNCcnt-synced).
// ---------------------------------------------------------------------------
__global__ __launch_bounds__(256) void attn_wmma_kernel(
    const __bf16* __restrict__ Q, const __bf16* __restrict__ Kb,
    const __bf16* __restrict__ Vt, __bf16* __restrict__ ctx, int T) {
  __shared__ __bf16 Ks[32 * 64];   // [s][d]
  __shared__ __bf16 VTs[64 * 32];  // [d][s]
  __shared__ __bf16 Ps[8][16 * 32];

  const int tid  = threadIdx.x;
  const int lane = tid & 31;
  const int wid  = tid >> 5;
  const int bh   = blockIdx.y;
  const int q0   = blockIdx.x * 128;
  const int qw   = q0 + wid * 16;

  const __bf16* Qb = Q  + (size_t)bh * T * 64;
  const __bf16* Kp = Kb + (size_t)bh * T * 64;
  const __bf16* Vp = Vt + (size_t)bh * 64 * T;

  bf16x16 qf[2];
#pragma unroll
  for (int c = 0; c < 2; ++c) qf[c] = frag_a(Qb + (size_t)qw * 64 + c * 32, 64, lane);

  v8f o[4];
#pragma unroll
  for (int j = 0; j < 4; ++j) o[j] = vzero8();
  float mi[8], li[8];
#pragma unroll
  for (int r = 0; r < 8; ++r) { mi[r] = -1e30f; li[r] = 0.f; }

  const int nl = lane & 15;
  const int kh = lane >> 4;

  for (int s0 = 0; s0 < q0 + 128; s0 += 32) {
    __syncthreads();
    {  // async K tile [32 s][64 d]: one B128 per thread
      int r = tid >> 3, kc = (tid & 7) * 8;
      async_copy_b128(Ks + r * 64 + kc, Kp + (size_t)(s0 + r) * 64 + kc);
    }
    {  // async V^T tile [64 d][32 s]
      int r = tid >> 2, sc = (tid & 3) * 8;
      async_copy_b128(VTs + r * 32 + sc, Vp + (size_t)r * T + s0 + sc);
    }
    wait_asynccnt0();
    __syncthreads();

    if (s0 <= qw + 15) {
      v8f s[2] = {vzero8(), vzero8()};
#pragma unroll
      for (int j = 0; j < 2; ++j)
#pragma unroll
        for (int c = 0; c < 2; ++c) {
          bf16x16 bk = frag_bt(Ks + (j * 16) * 64 + c * 32, 64, lane);
          s[j] = wmma_bf16(qf[c], bk, s[j]);
        }

#pragma unroll
      for (int r = 0; r < 8; ++r) {
        const int row = qw + r + 8 * kh;
        float v0 = (s0 + nl      <= row) ? s[0][r] : -1e30f;
        float v1 = (s0 + 16 + nl <= row) ? s[1][r] : -1e30f;
        float mv = fmaxf(v0, v1);
#pragma unroll
        for (int off = 1; off < 16; off <<= 1) mv = fmaxf(mv, __shfl_xor(mv, off));
        const float nm = fmaxf(mi[r], mv);
        const float pscale = __expf(mi[r] - nm);
        const float p0 = __expf(v0 - nm);
        const float p1 = __expf(v1 - nm);
        s[0][r] = p0;
        s[1][r] = p1;
        float ls = p0 + p1;
#pragma unroll
        for (int off = 1; off < 16; off <<= 1) ls += __shfl_xor(ls, off);
        li[r] = li[r] * pscale + ls;
        mi[r] = nm;
#pragma unroll
        for (int j = 0; j < 4; ++j) o[j][r] *= pscale;
      }

      __bf16* Pw = &Ps[wid][0];
#pragma unroll
      for (int r = 0; r < 8; ++r) {
        const int m = r + 8 * kh;
        Pw[m * 32 + nl]      = (__bf16)s[0][r];
        Pw[m * 32 + 16 + nl] = (__bf16)s[1][r];
      }
      bf16x16 pf = frag_a(Pw, 32, lane);
#pragma unroll
      for (int j = 0; j < 4; ++j) {
        bf16x16 bv = frag_bt(VTs + (j * 16) * 32, 32, lane);
        o[j] = wmma_bf16(pf, bv, o[j]);
      }
    }
  }

  const int b = bh >> 4;
  const int h = bh & 15;
#pragma unroll
  for (int j = 0; j < 4; ++j)
#pragma unroll
    for (int r = 0; r < 8; ++r) {
      const int t = qw + r + 8 * kh;
      const int d = j * 16 + nl;
      ctx[((size_t)b * T + t) * 1024 + h * 64 + d] = (__bf16)(o[j][r] / li[r]);
    }
}

// ---------------------------------------------------------------------------
// Host launcher
// ---------------------------------------------------------------------------
extern "C" void kernel_launch(void* const* d_in, const int* in_sizes, int n_in,
                              void* d_out, int out_size, void* d_ws, size_t ws_size,
                              hipStream_t stream) {
  const float* x   = (const float*)d_in[0];
  const float* Wq  = (const float*)d_in[1];
  const float* bq  = (const float*)d_in[2];
  const float* Wk  = (const float*)d_in[3];
  const float* bk  = (const float*)d_in[4];
  const float* Wv  = (const float*)d_in[5];
  const float* bv  = (const float*)d_in[6];
  const float* Wo  = (const float*)d_in[7];
  const float* bo  = (const float*)d_in[8];
  const float* W1  = (const float*)d_in[9];
  const float* b1  = (const float*)d_in[10];
  const float* W2  = (const float*)d_in[11];
  const float* b2  = (const float*)d_in[12];
  const float* g1  = (const float*)d_in[13];
  const float* be1 = (const float*)d_in[14];
  const float* g2  = (const float*)d_in[15];
  const float* be2 = (const float*)d_in[16];
  float* out = (float*)d_out;

  const int Bb = 2, T = 2048, C = 1024, H = 16, FF = 4096;
  const int M = Bb * T;  // 4096

  char* w = (char*)d_ws;
  auto alloc = [&](size_t bytes) {
    char* p = w;
    w += (bytes + 255) & ~(size_t)255;
    return p;
  };
  __bf16* ln1   = (__bf16*)alloc((size_t)M * C * 2);
  __bf16* wqkvT = (__bf16*)alloc((size_t)3 * C * C * 2);
  __bf16* woT   = (__bf16*)alloc((size_t)C * C * 2);
  __bf16* w1T   = (__bf16*)alloc((size_t)FF * C * 2);
  __bf16* w2T   = (__bf16*)alloc((size_t)C * FF * 2);
  __bf16* Qb    = (__bf16*)alloc((size_t)M * C * 2);
  __bf16* Kbf   = (__bf16*)alloc((size_t)M * C * 2);
  __bf16* Vtb   = (__bf16*)alloc((size_t)M * C * 2);
  __bf16* ctx   = (__bf16*)alloc((size_t)M * C * 2);
  float*  x2    = (float*)alloc((size_t)M * C * 4);
  __bf16* ln2   = (__bf16*)alloc((size_t)M * C * 2);
  __bf16* ff1   = (__bf16*)alloc((size_t)M * FF * 2);

  transpose_cvt_kernel<<<(C * C + 255) / 256, 256, 0, stream>>>(Wq, wqkvT + 0 * (size_t)C * C, C, C);
  transpose_cvt_kernel<<<(C * C + 255) / 256, 256, 0, stream>>>(Wk, wqkvT + 1 * (size_t)C * C, C, C);
  transpose_cvt_kernel<<<(C * C + 255) / 256, 256, 0, stream>>>(Wv, wqkvT + 2 * (size_t)C * C, C, C);
  transpose_cvt_kernel<<<(C * C + 255) / 256, 256, 0, stream>>>(Wo, woT, C, C);
  transpose_cvt_kernel<<<(C * FF + 255) / 256, 256, 0, stream>>>(W1, w1T, C, FF);
  transpose_cvt_kernel<<<(FF * C + 255) / 256, 256, 0, stream>>>(W2, w2T, FF, C);

  layernorm_bf16_kernel<<<M, 256, 0, stream>>>(x, g1, be1, ln1);

  {
    dim3 grid(3 * C / 128, M / 128);
    gemm_wmma_kernel<3><<<grid, 256, 0, stream>>>(ln1, wqkvT, M, 3 * C, C,
        bq, bk, bv, nullptr, nullptr, Qb, Kbf, Vtb, T);
  }
  {
    dim3 grid(T / 128, Bb * H);
    attn_wmma_kernel<<<grid, 256, 0, stream>>>(Qb, Kbf, Vtb, ctx, T);
  }
  {
    dim3 grid(C / 128, M / 128);
    gemm_wmma_kernel<2><<<grid, 256, 0, stream>>>(ctx, woT, M, C, C,
        bo, nullptr, nullptr, x, x2, nullptr, nullptr, nullptr, T);
  }
  layernorm_bf16_kernel<<<M, 256, 0, stream>>>(x2, g2, be2, ln2);
  {
    dim3 grid(FF / 128, M / 128);
    gemm_wmma_kernel<1><<<grid, 256, 0, stream>>>(ln2, w1T, M, FF, C,
        b1, nullptr, nullptr, nullptr, nullptr, ff1, nullptr, nullptr, T);
  }
  {
    dim3 grid(C / 128, M / 128);
    gemm_wmma_kernel<2><<<grid, 256, 0, stream>>>(ff1, w2T, M, C, FF,
        b2, nullptr, nullptr, x2, out, nullptr, nullptr, nullptr, T);
  }
}
